// GRAN_62122406969869
// MI455X (gfx1250) — matrix operand order repaired and checked by reading, not compiled, not run on hardware
//
#include <hip/hip_runtime.h>
#include <math.h>

// ---------------- CDNA5 types ----------------
typedef __attribute__((ext_vector_type(16))) __bf16 v16bf;
typedef __attribute__((ext_vector_type(8)))  float  v8f;
typedef __attribute__((ext_vector_type(2)))  __bf16 bf2;

// ---------------- constants ----------------
#define Bt   128
#define St   64
#define FIN  32
#define Et   96
#define E2t  192
#define Ht   8
#define Dt   24
#define Lt   4
#define HORt 24
#define FLATt (St*E2t)          // 12288
#define ROWS  (Bt*St)           // 8192
#define KCONV (4*4*Et)          // 1536

// ---------------- helpers ----------------
// native f32->bf16 convert (RNE; lowers to v_cvt_*bf16_f32 on gfx1250)
__device__ __forceinline__ __bf16 f2bf(float f) { return (__bf16)f; }

__device__ __forceinline__ unsigned pack2bf(float lo, float hi) {
    bf2 p = { (__bf16)lo, (__bf16)hi };
    return __builtin_bit_cast(unsigned, p);
}

__device__ __forceinline__ float gelu_f(float x) {
    const float c = 0.7978845608028654f;           // sqrt(2/pi)
    float t = tanhf(c * (x + 0.044715f * x * x * x));
    return 0.5f * x * (1.0f + t);
}

__device__ __forceinline__ float softplus_f(float x) {
    return (x > 20.0f) ? x : log1pf(expf(x));
}

// LDS offset (wave-relative) from a flat shared pointer: low 32 bits of the
// shared-aperture flat address (ISA 10.2).
__device__ __forceinline__ unsigned lds_off(const void* p) {
    return (unsigned)(unsigned long long)p;
}

// ================= proj: gelu(LN(x @ W + b)) =================
__global__ void proj_kernel(const float* __restrict__ x,
                            const float* __restrict__ W,
                            const float* __restrict__ b,
                            const float* __restrict__ lns,
                            const float* __restrict__ lno,
                            float* __restrict__ h0) {
    int row = blockIdx.x;
    int t = threadIdx.x;
    __shared__ float sx[FIN];
    __shared__ float red[128];
    if (t < FIN) sx[t] = x[row * FIN + t];
    __syncthreads();
    float val = 0.0f;
    if (t < Et) {
        float acc = b[t];
        #pragma unroll
        for (int k = 0; k < FIN; ++k) acc += sx[k] * W[k * Et + t];
        val = acc;
    }
    red[t] = (t < Et) ? val : 0.0f;
    __syncthreads();
    for (int o2 = 64; o2 > 0; o2 >>= 1) { if (t < o2) red[t] += red[t + o2]; __syncthreads(); }
    float mu = red[0] / (float)Et;
    __syncthreads();
    float d = (t < Et) ? (val - mu) : 0.0f;
    red[t] = d * d;
    __syncthreads();
    for (int o2 = 64; o2 > 0; o2 >>= 1) { if (t < o2) red[t] += red[t + o2]; __syncthreads(); }
    float var = red[0] / (float)Et;
    if (t < Et) {
        float y = (val - mu) * rsqrtf(var + 1e-5f) * lns[t] + lno[t];
        h0[row * Et + t] = gelu_f(y);
    }
}

// ================= im2col (4x4 SAME, pad lo=1 hi=2) -> bf16 =================
__global__ void im2col_kernel(const float* __restrict__ h0, __bf16* __restrict__ out) {
    int idx = blockIdx.x * 256 + threadIdx.x;
    if (idx >= ROWS * KCONV) return;
    int k = idx % KCONV, p = idx / KCONV;
    int s = p % St, bb = p / St;
    int ci = k % Et, r = k / Et;
    int dw = r & 3, dh = r >> 2;
    int ih = bb + dh - 1, iw = s + dw - 1;
    float v = 0.0f;
    if (ih >= 0 && ih < Bt && iw >= 0 && iw < St) v = h0[(ih * St + iw) * Et + ci];
    out[idx] = f2bf(v);
}

// ================= generic WMMA GEMM =================
// C[M,N] = epi( A_bf16[M,K] @ W_f32[K,N] + bias )    epi: 0=none 1=gelu 2=+resid
// Tile 128M x 64N x 32K. 8 waves; wave w owns rows [16w,16w+16) x all 64 N
// -> 4 accumulators, 4 v_wmma per K-step. Double-buffered LDS, 1 barrier/step.
// A tile: global_load_async_to_lds_b128 DMA (ASYNCcnt-fenced).
// W tile: float4 register staging (coalesced along N), v_cvt pack to bf16, b32 stores.
__global__ __launch_bounds__(256)
void gemm_bf16_wmma_kernel(const __bf16* __restrict__ A, int lda,
                           const float* __restrict__ W, int ldw,
                           const float* __restrict__ bias,
                           const float* __restrict__ resid,
                           float* __restrict__ C, int ldc,
                           int M, int N, int K, int epi) {
    __shared__ uint4  sA4[2][512];       // 128x32 bf16, [m][k]: idx = m*4 + k/8
    __shared__ __bf16 sB[2][64 * 32];    // [n][k] (K contiguous -> b128 frag loads)

    int tid  = threadIdx.x;
    int lane = tid & 31;
    int wave = tid >> 5;
    int m0 = blockIdx.y * 128;
    int n0 = blockIdx.x * 64;
    int row = lane & 15;
    int hi  = (lane >> 4) & 1;

    v8f acc[4];
    #pragma unroll
    for (int i = 0; i < 4; ++i) acc[i] = (v8f){};

    // A staging via async DMA: 512 uint4 chunks, 2 per thread
    int ca0 = tid, ca1 = tid + 256;
    int am0 = ca0 >> 2, ak0 = (ca0 & 3) * 8;
    int am1 = ca1 >> 2, ak1 = (ca1 & 3) * 8;
    // W staging: thread covers adjacent k-pair (kp, kp+1) x 4 consecutive n
    int kp = (tid >> 4) * 2;      // 0,2,..,30
    int wn = (tid & 15) * 4;      // 0..60

    float4 rw0, rw1;

    auto issue_async_a = [&](int k0, int buf) {
        unsigned l0 = lds_off(&sA4[buf][ca0]);
        unsigned l1 = lds_off(&sA4[buf][ca1]);
        unsigned long long g0 =
            (unsigned long long)(const void*)(A + (size_t)(m0 + am0) * lda + k0 + ak0);
        unsigned long long g1 =
            (unsigned long long)(const void*)(A + (size_t)(m0 + am1) * lda + k0 + ak1);
        asm volatile("global_load_async_to_lds_b128 %0, %1, off"
                     :: "v"(l0), "v"(g0) : "memory");
        asm volatile("global_load_async_to_lds_b128 %0, %1, off"
                     :: "v"(l1), "v"(g1) : "memory");
    };
    auto load_w = [&](int k0) {
        rw0 = *(const float4*)(W + (size_t)(k0 + kp) * ldw + n0 + wn);
        rw1 = *(const float4*)(W + (size_t)(k0 + kp + 1) * ldw + n0 + wn);
    };
    auto store_w = [&](int buf) {
        __bf16* b = &sB[buf][0];
        *(unsigned*)&b[(wn + 0) * 32 + kp] = pack2bf(rw0.x, rw1.x);
        *(unsigned*)&b[(wn + 1) * 32 + kp] = pack2bf(rw0.y, rw1.y);
        *(unsigned*)&b[(wn + 2) * 32 + kp] = pack2bf(rw0.z, rw1.z);
        *(unsigned*)&b[(wn + 3) * 32 + kp] = pack2bf(rw0.w, rw1.w);
    };

    int nk = K >> 5;
    load_w(0);
    issue_async_a(0, 0);
    store_w(0);
    asm volatile("s_wait_asynccnt 0x0" ::: "memory");

    for (int it = 0; it < nk; ++it) {
        __syncthreads();                       // lds[it&1] ready; lds[(it+1)&1] free
        int buf = it & 1;
        if (it + 1 < nk) {
            load_w((it + 1) << 5);             // f32 W regs for next tile
            issue_async_a((it + 1) << 5, buf ^ 1);
        }
        if (it + 2 < nk)                       // hint tile it+2 (global_prefetch_b8)
            __builtin_prefetch(W + (size_t)(((it + 2) << 5) + kp) * ldw + n0 + wn, 0, 1);

        const __bf16* a = (const __bf16*)&sA4[buf][0];

        v16bf afrag;                           // 16-bit A 16x32 lane layout (ISA 7.12.2)
        #pragma unroll
        for (int v = 0; v < 8; ++v) {
            int kk = (hi ? 8 : 0) + ((v >= 4) ? 16 : 0) + 2 * (v & 3);
            int base = (wave * 16 + row) * 32 + kk;
            afrag[2 * v]     = a[base];
            afrag[2 * v + 1] = a[base + 1];
        }
        v16bf bfr[4];
        #pragma unroll
        for (int ns = 0; ns < 4; ++ns) {
            #pragma unroll
            for (int v = 0; v < 8; ++v) {
                int kk = (hi ? 8 : 0) + ((v >= 4) ? 16 : 0) + 2 * (v & 3);
                int base = (ns * 16 + row) * 32 + kk;
                bfr[ns][2 * v]     = sB[buf][base];
                bfr[ns][2 * v + 1] = sB[buf][base + 1];
            }
        }
        #pragma unroll
        for (int ns = 0; ns < 4; ++ns) {
            acc[ns] = __builtin_amdgcn_wmma_f32_16x16x32_bf16(false, afrag, false, bfr[ns],
                                                              (short)0, acc[ns], false, false);
        }
        if (it + 1 < nk) {
            store_w(buf ^ 1);
            asm volatile("s_wait_asynccnt 0x0" ::: "memory");
        }
    }

    // C/D layout: lanes 0-15 -> M=r, lanes 16-31 -> M=8+r; N = lane&15
    int cmb = wave * 16 + (hi ? 8 : 0);
    int cn  = lane & 15;
    #pragma unroll
    for (int r = 0; r < 8; ++r) {
        int mg = m0 + cmb + r;
        #pragma unroll
        for (int ns = 0; ns < 4; ++ns) {
            int ng = n0 + ns * 16 + cn;
            float v = acc[ns][r] + bias[ng];
            if (epi == 1) v = gelu_f(v);
            else if (epi == 2) v += resid[(size_t)mg * ldc + ng];
            C[(size_t)mg * ldc + ng] = v;
        }
    }
}

// ================= row LayerNorm (+opt residual, +opt gelu, +opt bf16 copy) ====
__global__ void row_ln_kernel(const float* __restrict__ X,
                              const float* __restrict__ R,
                              const float* __restrict__ sc,
                              const float* __restrict__ of,
                              float* __restrict__ Y,
                              __bf16* __restrict__ Ybf,
                              int N, int gelu_after) {
    int row = blockIdx.x;
    int t = threadIdx.x;
    const float* xr = X + (size_t)row * N;
    const float* rr = R ? (R + (size_t)row * N) : nullptr;
    __shared__ float red[256];
    float s1 = 0.0f, s2 = 0.0f;
    for (int i = t; i < N; i += 256) {
        float v = xr[i] + (rr ? rr[i] : 0.0f);
        s1 += v; s2 += v * v;
    }
    red[t] = s1; __syncthreads();
    for (int o2 = 128; o2 > 0; o2 >>= 1) { if (t < o2) red[t] += red[t + o2]; __syncthreads(); }
    s1 = red[0]; __syncthreads();
    red[t] = s2; __syncthreads();
    for (int o2 = 128; o2 > 0; o2 >>= 1) { if (t < o2) red[t] += red[t + o2]; __syncthreads(); }
    s2 = red[0]; __syncthreads();
    float mu  = s1 / (float)N;
    float var = s2 / (float)N - mu * mu;
    float inv = rsqrtf(var + 1e-5f);
    for (int i = t; i < N; i += 256) {
        float v = xr[i] + (rr ? rr[i] : 0.0f);
        float y = (v - mu) * inv * sc[i] + of[i];
        if (gelu_after) y = gelu_f(y);
        Y[(size_t)row * N + i] = y;
        if (Ybf) Ybf[(size_t)row * N + i] = f2bf(y);
    }
}

// ================= mass = softplus(h @ mw + mb)  (8192 x 8) =================
__global__ void mass_kernel(const float* __restrict__ h,
                            const float* __restrict__ mw,
                            const float* __restrict__ mb,
                            float* __restrict__ mass) {
    int idx = blockIdx.x * 256 + threadIdx.x;
    if (idx >= ROWS * Ht) return;
    int row = idx >> 3, j = idx & 7;
    const float* hr = h + (size_t)row * E2t;
    float acc = mb[j];
    for (int k = 0; k < E2t; ++k) acc += hr[k] * mw[k * Ht + j];
    mass[idx] = softplus_f(acc);
}

// ================= gravity attention, one WG per (b,head), 64 threads =================
__global__ void attn_kernel(const float* __restrict__ qk,
                            const float* __restrict__ vv,
                            const float* __restrict__ mass,
                            float* __restrict__ aout) {
    int b  = blockIdx.x >> 3;
    int hd = blockIdx.x & 7;
    __shared__ float sq[St * Dt];
    __shared__ float sv[St * Dt];
    __shared__ float sm[St];
    __shared__ float ss[St];
    int t = threadIdx.x;   // 64
    for (int i = t; i < St * Dt; i += 64) {
        int s = i / Dt, d = i % Dt;
        size_t base = (size_t)(b * St + s) * E2t + hd * Dt + d;
        sq[i] = qk[base];
        sv[i] = vv[base];
    }
    sm[t] = mass[(b * St + t) * Ht + hd];
    __syncthreads();
    {
        float a = 0.0f;
        #pragma unroll
        for (int d = 0; d < Dt; ++d) { float q = sq[t * Dt + d]; a += q * q; }
        ss[t] = a;
    }
    __syncthreads();
    int i = t;
    float mi = sm[i], sqi = ss[i];
    float mx = -1e30f;
    for (int j = 0; j < St; ++j) {
        float dot = 0.0f;
        #pragma unroll
        for (int d = 0; d < Dt; ++d) dot += sq[i * Dt + d] * sq[j * Dt + d];
        float d2 = fmaxf(sqi + ss[j] - 2.0f * dot, 0.0f) + 1e-6f;
        float scv = mi * sm[j] / d2;
        mx = fmaxf(mx, scv);
    }
    float sum = 0.0f;
    float outr[Dt];
    #pragma unroll
    for (int d = 0; d < Dt; ++d) outr[d] = 0.0f;
    for (int j = 0; j < St; ++j) {
        float dot = 0.0f;
        #pragma unroll
        for (int d = 0; d < Dt; ++d) dot += sq[i * Dt + d] * sq[j * Dt + d];
        float d2 = fmaxf(sqi + ss[j] - 2.0f * dot, 0.0f) + 1e-6f;
        float w = expf(mi * sm[j] / d2 - mx);
        sum += w;
        #pragma unroll
        for (int d = 0; d < Dt; ++d) outr[d] += w * sv[j * Dt + d];
    }
    float inv = 1.0f / sum;
    size_t obase = (size_t)(b * St + i) * E2t + hd * Dt;
    #pragma unroll
    for (int d = 0; d < Dt; ++d) aout[obase + d] = outr[d] * inv;
}

// ================= f32 -> bf16 convert =================
__global__ void f2bf_kernel(const float* __restrict__ X, __bf16* __restrict__ Y, int n) {
    int i = blockIdx.x * 256 + threadIdx.x;
    if (i < n) Y[i] = f2bf(X[i]);
}

// ================= final: out = f @ out_w + out_b  (128 x 24, K=12288) =================
__global__ void out_kernel(const float* __restrict__ f,
                           const float* __restrict__ W,
                           const float* __restrict__ b,
                           float* __restrict__ o) {
    int bi = blockIdx.x;
    int t = threadIdx.x;   // 256
    __shared__ float red[240];
    if (t < 240) {
        int j = t % HORt, g = t / HORt;   // g 0..9
        const float* fr = f + (size_t)bi * FLATt;
        float acc = 0.0f;
        for (int k = g; k < FLATt; k += 10) acc += fr[k] * W[(size_t)k * HORt + j];
        red[t] = acc;
    }
    __syncthreads();
    if (t < HORt) {
        float a = b[t];
        for (int g = 0; g < 10; ++g) a += red[g * HORt + t];
        o[bi * HORt + t] = a;
    }
}

// ================= host orchestration =================
extern "C" void kernel_launch(void* const* d_in, const int* in_sizes, int n_in,
                              void* d_out, int out_size, void* d_ws, size_t ws_size,
                              hipStream_t stream) {
    (void)in_sizes; (void)n_in; (void)out_size; (void)ws_size;
    const float* x      = (const float*)d_in[0];
    const float* proj_w = (const float*)d_in[1];
    const float* proj_b = (const float*)d_in[2];
    const float* ln0_s  = (const float*)d_in[3];
    const float* ln0_o  = (const float*)d_in[4];
    const float* conv_w = (const float*)d_in[5];
    const float* conv_b = (const float*)d_in[6];
    const float* ln1_s  = (const float*)d_in[7];
    const float* ln1_o  = (const float*)d_in[8];
    const float* qk_w   = (const float*)d_in[9];
    const float* qk_b   = (const float*)d_in[10];
    const float* mass_w = (const float*)d_in[11];
    const float* mass_b = (const float*)d_in[12];
    const float* v_w    = (const float*)d_in[13];
    const float* v_b    = (const float*)d_in[14];
    const float* norm_s = (const float*)d_in[15];
    const float* norm_o = (const float*)d_in[16];
    const float* mlp_w1 = (const float*)d_in[17];
    const float* mlp_b1 = (const float*)d_in[18];
    const float* mlp_w2 = (const float*)d_in[19];
    const float* mlp_b2 = (const float*)d_in[20];
    const float* fc_w   = (const float*)d_in[21];
    const float* fc_b   = (const float*)d_in[22];
    const float* ln2_s  = (const float*)d_in[23];
    const float* ln2_o  = (const float*)d_in[24];
    const float* out_w  = (const float*)d_in[25];
    const float* out_b  = (const float*)d_in[26];
    float* out = (float*)d_out;

    // ---- workspace arena (256B aligned blocks) ----
    char* ws = (char*)d_ws;
    size_t off = 0;
    auto alloc = [&](size_t bytes) -> char* {
        char* p = ws + off;
        off += (bytes + 255) & ~(size_t)255;
        return p;
    };
    float*  h0    = (float*) alloc((size_t)ROWS * Et * 4);
    __bf16* im2   = (__bf16*)alloc((size_t)ROWS * KCONV * 2);
    float*  h1    = (float*) alloc((size_t)ROWS * E2t * 4);
    __bf16* h1bf  = (__bf16*)alloc((size_t)ROWS * E2t * 2);
    float*  bufA  = (float*) alloc((size_t)ROWS * 4 * Et * 4);   // conv-y / mlp-t1
    __bf16* t1bf  = (__bf16*)alloc((size_t)ROWS * 4 * Et * 2);
    float*  qkb   = (float*) alloc((size_t)ROWS * E2t * 4);
    float*  vvb   = (float*) alloc((size_t)ROWS * E2t * 4);
    float*  msb   = (float*) alloc((size_t)ROWS * Ht * 4);
    float*  aoutb = (float*) alloc((size_t)ROWS * E2t * 4);
    float*  fpre  = (float*) alloc((size_t)Bt * FLATt * 4);
    float*  fbuf  = (float*) alloc((size_t)Bt * FLATt * 4);

    // 1. proj -> LN -> gelu
    proj_kernel<<<ROWS, 128, 0, stream>>>(x, proj_w, proj_b, ln0_s, ln0_o, h0);

    // 2. im2col (bf16) + conv as GEMM: M=8192 K=1536 N=192
    {
        int tot = ROWS * KCONV;
        im2col_kernel<<<(tot + 255) / 256, 256, 0, stream>>>(h0, im2);
        dim3 g(E2t / 64, ROWS / 128);
        gemm_bf16_wmma_kernel<<<g, 256, 0, stream>>>(im2, KCONV, conv_w, E2t, conv_b,
                                                     nullptr, bufA, E2t,
                                                     ROWS, E2t, KCONV, 0);
    }
    // 3. LN1 + gelu
    row_ln_kernel<<<ROWS, 256, 0, stream>>>(bufA, nullptr, ln1_s, ln1_o, h1, h1bf, E2t, 1);

    // 4. L gravity-attention layers
    for (int l = 0; l < Lt; ++l) {
        dim3 g(E2t / 64, ROWS / 128);
        gemm_bf16_wmma_kernel<<<g, 256, 0, stream>>>(h1bf, E2t, qk_w + (size_t)l * E2t * E2t,
                                                     E2t, qk_b + (size_t)l * E2t,
                                                     nullptr, qkb, E2t, ROWS, E2t, E2t, 0);
        gemm_bf16_wmma_kernel<<<g, 256, 0, stream>>>(h1bf, E2t, v_w + (size_t)l * E2t * E2t,
                                                     E2t, v_b + (size_t)l * E2t,
                                                     nullptr, vvb, E2t, ROWS, E2t, E2t, 0);
        mass_kernel<<<(ROWS * Ht + 255) / 256, 256, 0, stream>>>(
            h1, mass_w + (size_t)l * E2t * Ht, mass_b + (size_t)l * Ht, msb);
        attn_kernel<<<Bt * Ht, 64, 0, stream>>>(qkb, vvb, msb, aoutb);
        row_ln_kernel<<<ROWS, 256, 0, stream>>>(h1, aoutb, norm_s + (size_t)l * E2t,
                                                norm_o + (size_t)l * E2t, h1, h1bf, E2t, 1);
    }

    // 5. MLP: h += gelu(h@w1+b1)@w2+b2
    {
        dim3 g1(4 * Et / 64, ROWS / 128);
        gemm_bf16_wmma_kernel<<<g1, 256, 0, stream>>>(h1bf, E2t, mlp_w1, 4 * Et, mlp_b1,
                                                      nullptr, bufA, 4 * Et,
                                                      ROWS, 4 * Et, E2t, 1);
        int n1 = ROWS * 4 * Et;
        f2bf_kernel<<<(n1 + 255) / 256, 256, 0, stream>>>(bufA, t1bf, n1);
        dim3 g2(E2t / 64, ROWS / 128);
        gemm_bf16_wmma_kernel<<<g2, 256, 0, stream>>>(t1bf, 4 * Et, mlp_w2, E2t, mlp_b2,
                                                      h1, h1, E2t, ROWS, E2t, 4 * Et, 2);
        int n2 = ROWS * E2t;
        f2bf_kernel<<<(n2 + 255) / 256, 256, 0, stream>>>(h1, h1bf, n2);
    }

    // 6. fc: (128 x 12288) @ (12288 x 12288), gelu epilogue
    //    grid (192,1): each fc_w element streamed from HBM exactly once (~604MB @23.3TB/s)
    {
        dim3 g(FLATt / 64, Bt / 128);
        gemm_bf16_wmma_kernel<<<g, 256, 0, stream>>>(h1bf, FLATt, fc_w, FLATt, fc_b,
                                                     nullptr, fpre, FLATt,
                                                     Bt, FLATt, FLATt, 1);
    }
    // 7. LN2
    row_ln_kernel<<<Bt, 256, 0, stream>>>(fpre, nullptr, ln2_s, ln2_o, fbuf, nullptr, FLATt, 0);

    // 8. out head
    out_kernel<<<Bt, 256, 0, stream>>>(fbuf, out_w, out_b, out);
}
